// TemporalTransformer_69801808495150
// MI455X (gfx1250) — matrix operand order repaired and checked by reading, not compiled, not run on hardware
//
#include <hip/hip_runtime.h>
#include <cstdint>
#include <cstddef>

// ---------------------------------------------------------------- types ----
typedef __attribute__((ext_vector_type(16))) __bf16 v16bf;
typedef __attribute__((ext_vector_type(8)))  float  v8f;
typedef __attribute__((ext_vector_type(4)))  int    v4i;

#define D_MODEL 512
#define NHEAD   8
#define HDIM    64
#define BATCH   8
#define AS      40   // padded LDS row stride (bf16 elems) for 32-wide K tiles

#if defined(__has_builtin)
#  if __has_builtin(__builtin_amdgcn_global_load_async_to_lds_b128) && \
      __has_builtin(__builtin_amdgcn_s_wait_asynccnt)
#    define HAS_ASYNC 1
#  endif
#endif
#ifndef HAS_ASYNC
#  define HAS_ASYNC 0
#endif

// f32 -> bf16, round-to-nearest-even
__device__ __forceinline__ unsigned short f2bf(float f) {
    unsigned int u = __float_as_uint(f);
    unsigned int r = u + 0x7FFFu + ((u >> 16) & 1u);
    return (unsigned short)(r >> 16);
}

// 16B global->LDS copy: async DMA (ASYNCcnt) when available, else sync b128.
__device__ __forceinline__ void cp16(const unsigned short* g, unsigned short* l) {
#if HAS_ASYNC
    __builtin_amdgcn_global_load_async_to_lds_b128(
        (__attribute__((address_space(1))) v4i*)g,
        (__attribute__((address_space(3))) v4i*)l, 0, 0);
#else
    *(uint4*)l = *(const uint4*)g;
#endif
}
__device__ __forceinline__ void cp_wait() {
#if HAS_ASYNC
    __builtin_amdgcn_s_wait_asynccnt(0);
#endif
}

// A fragment: per lane elems [kb..kb+7] and [kb+16..kb+23]
__device__ __forceinline__ v16bf ld_frag_gap(const unsigned short* p) {
    union { uint4 q[2]; v16bf v; } t;
    t.q[0] = *(const uint4*)(p);
    t.q[1] = *(const uint4*)(p + 16);
    return t.v;
}
// Unused (B uses ld_frag_gap): per lane 16 contiguous K elems (half-wave K split)
__device__ __forceinline__ v16bf ld_frag_seq(const unsigned short* p) {
    union { uint4 q[2]; v16bf v; } t;
    t.q[0] = *(const uint4*)(p);
    t.q[1] = *(const uint4*)(p + 8);
    return t.v;
}

// One 16(M)x64(N)x32(K) step per wave: 4 chained v_wmma_f32_16x16x32_bf16
__device__ __forceinline__ void wmma_step(const unsigned short* sA,
                                          const unsigned short* sB,
                                          int wave, int lane, v8f acc[4]) {
    const int m  = lane & 15;
    const int kb = (lane >> 4) << 3;   // A: 0 or 8
    const int kh = (lane >> 4) << 4;   // unused (B uses kb)
    v16bf af = ld_frag_gap(sA + (wave * 16 + m) * AS + kb);
#pragma unroll
    for (int nt = 0; nt < 4; ++nt) {
        v16bf bf = ld_frag_gap(sB + (nt * 16 + m) * AS + kb);
        acc[nt] = __builtin_amdgcn_wmma_f32_16x16x32_bf16(
            false, af, false, bf, (short)0, acc[nt], false, false);
    }
}

template <bool RELU, bool OUTBF>
__device__ __forceinline__ void store_tile(void* __restrict__ Cout,
                                           const float* __restrict__ bias,
                                           const v8f acc[4],
                                           int bm, int bn, int N,
                                           int wave, int lane) {
    const int rbase = bm + wave * 16 + ((lane >> 4) << 3);
    const int cl    = lane & 15;
#pragma unroll
    for (int nt = 0; nt < 4; ++nt) {
        const int col = bn + nt * 16 + cl;
        const float bv = bias ? bias[col] : 0.f;
#pragma unroll
        for (int r = 0; r < 8; ++r) {
            float v = acc[nt][r] + bv;
            if (RELU) v = fmaxf(v, 0.f);
            if (OUTBF) ((unsigned short*)Cout)[(size_t)(rbase + r) * N + col] = f2bf(v);
            else       ((float*)Cout)[(size_t)(rbase + r) * N + col] = v;
        }
    }
}

// ------------------------------------------------- all-bf16 WMMA GEMM ----
// C[M,N] = A[M,K] @ Bt[N,K]^T + bias.  A row-major bf16, Bt N-major bf16.
// Double-buffered LDS staging via async global->LDS b128.
template <bool RELU, bool OUTBF>
__global__ __launch_bounds__(128) void gemm_bf16(
    const unsigned short* __restrict__ A, const unsigned short* __restrict__ Bt,
    const float* __restrict__ bias, void* __restrict__ Cout,
    int M, int N, int K) {
    __shared__ __align__(16) unsigned short sA[2][64 * AS];
    __shared__ __align__(16) unsigned short sB[2][64 * AS];
    const int tid  = threadIdx.x;
    const int wave = tid >> 5, lane = tid & 31;
    const int bm = blockIdx.y * 64, bn = blockIdx.x * 64;
    v8f acc[4] = {};

    auto issue = [&](int buf, int k0) {
#pragma unroll
        for (int i = 0; i < 2; ++i) {           // A tile: 64 rows x 32 bf16
            const int slot = tid + i * 128;     // 256 x 16B transfers
            const int row = slot >> 2, c8 = slot & 3;
            cp16(A + (size_t)(bm + row) * K + k0 + c8 * 8,
                 &sA[buf][row * AS + c8 * 8]);
        }
#pragma unroll
        for (int i = 0; i < 2; ++i) {           // B tile: 64 n-rows x 32 bf16
            const int slot = tid + i * 128;
            const int row = slot >> 2, c8 = slot & 3;
            cp16(Bt + (size_t)(bn + row) * K + k0 + c8 * 8,
                 &sB[buf][row * AS + c8 * 8]);
        }
    };

    issue(0, 0);
    int cur = 0;
    for (int k0 = 0; k0 < K; k0 += 32) {
        cp_wait();
        __syncthreads();
        if (k0 + 32 < K) issue(cur ^ 1, k0 + 32);   // overlap next stage with math
        wmma_step(sA[cur], sB[cur], wave, lane, acc);
        cur ^= 1;
#if !HAS_ASYNC
        __syncthreads();
#endif
    }
    store_tile<RELU, OUTBF>(Cout, bias, acc, bm, bn, N, wave, lane);
}

// ---------------------------------------------- k=3 conv as im2col WMMA ----
// Y[b,l,n] = bias[n] + sum_{tap,din} X[b,l+tap-padOff,din] * W[n,din,tap]
// Xbf: (B,Lc,D) bf16; Wt: (D x 1536) bf16, Wt[n][tap*512+din].
__global__ __launch_bounds__(128) void conv3_bf16(
    const unsigned short* __restrict__ Xbf, const unsigned short* __restrict__ Wt,
    const float* __restrict__ bias, float* __restrict__ Y,
    int Lc, int lshift, int padOff) {
    __shared__ __align__(16) unsigned short sA[64 * AS];
    __shared__ __align__(16) unsigned short sB[64 * AS];
    const int tid  = threadIdx.x;
    const int wave = tid >> 5, lane = tid & 31;
    const int bm = blockIdx.y * 64, bn = blockIdx.x * 64;
    v8f acc[4] = {};
    for (int k0 = 0; k0 < 3 * D_MODEL; k0 += 32) {
        const int tap = k0 >> 9, din0 = k0 & 511;   // constant within a chunk
#pragma unroll
        for (int i = 0; i < 2; ++i) {               // B tile via async DMA
            const int slot = tid + i * 128;
            const int row = slot >> 2, c8 = slot & 3;
            cp16(Wt + (size_t)(bn + row) * (3 * D_MODEL) + k0 + c8 * 8,
                 &sB[row * AS + c8 * 8]);
        }
#pragma unroll
        for (int i = 0; i < 2; ++i) {               // im2col A tile (bounds -> 0)
            const int slot = tid + i * 128;
            const int row = slot >> 2, c8 = slot & 3;
            const int gt = bm + row;
            const int b = gt >> lshift, l = gt & (Lc - 1);
            const int lsrc = l + tap - padOff;
            uint4 v = make_uint4(0u, 0u, 0u, 0u);
            if ((unsigned)lsrc < (unsigned)Lc)
                v = *(const uint4*)(Xbf + (((size_t)b * Lc + lsrc) << 9) + din0 + c8 * 8);
            *(uint4*)&sA[row * AS + c8 * 8] = v;
        }
        cp_wait();
        __syncthreads();
        wmma_step(sA, sB, wave, lane, acc);
        __syncthreads();
    }
    store_tile<false, false>(Y, bias, acc, bm, bn, D_MODEL, wave, lane);
}

// ------------------------------------------------------ weight converts ----
__global__ __launch_bounds__(256) void cvt_bf16_kernel(
    const float* __restrict__ src, unsigned short* __restrict__ dst, int n) {
    const int i = blockIdx.x * 256 + threadIdx.x;
    if (i < n) dst[i] = f2bf(src[i]);
}
// Wt[e][n][k] = W[e][k][n]
__global__ __launch_bounds__(256) void transpose_cvt_kernel(
    const float* __restrict__ W, unsigned short* __restrict__ Wt,
    int K, int N, int total) {
    const int o = blockIdx.x * 256 + threadIdx.x;
    if (o >= total) return;
    const int kn = K * N;
    const int e = o / kn, r = o - e * kn;
    const int n = r / K, k = r - n * K;
    Wt[o] = f2bf(W[(size_t)e * kn + (size_t)k * N + n]);
}
// Wt[e][n][tap*512+din] = W[e][n][din][tap]   (conv weights (E,D,D,3))
__global__ __launch_bounds__(256) void conv_cvt_kernel(
    const float* __restrict__ W, unsigned short* __restrict__ Wt, int total) {
    const int o = blockIdx.x * 256 + threadIdx.x;
    if (o >= total) return;
    const int en = o / 1536;                 // e*512 + n
    const int r = o - en * 1536;
    const int tap = r >> 9, din = r & 511;
    Wt[o] = f2bf(W[(size_t)en * 1536 + din * 3 + tap]);
}

// ------------------------------------------------- LayerNorm(residual) ----
__global__ __launch_bounds__(256) void ln_kernel(
    const float* __restrict__ x, const float* __restrict__ r,
    const float* __restrict__ g, const float* __restrict__ bta,
    float* __restrict__ out, unsigned short* __restrict__ outbf) {
    __shared__ float red[256];
    const int row = blockIdx.x, tid = threadIdx.x;
    const size_t o = (size_t)row * D_MODEL;
    const float v0 = x[o + tid]       + r[o + tid];
    const float v1 = x[o + 256 + tid] + r[o + 256 + tid];
    red[tid] = v0 + v1; __syncthreads();
    for (int k = 128; k > 0; k >>= 1) { if (tid < k) red[tid] += red[tid + k]; __syncthreads(); }
    const float mu = red[0] * (1.f / D_MODEL); __syncthreads();
    const float c0 = v0 - mu, c1 = v1 - mu;
    red[tid] = c0 * c0 + c1 * c1; __syncthreads();
    for (int k = 128; k > 0; k >>= 1) { if (tid < k) red[tid] += red[tid + k]; __syncthreads(); }
    const float rs = rsqrtf(red[0] * (1.f / D_MODEL) + 1e-5f);
    const float o0 = g[tid]       * c0 * rs + bta[tid];
    const float o1 = g[tid + 256] * c1 * rs + bta[tid + 256];
    out[o + tid]       = o0;  outbf[o + tid]       = f2bf(o0);
    out[o + 256 + tid] = o1;  outbf[o + 256 + tid] = f2bf(o1);
}

// ------------------------------------------------- embedding epilogue ----
__global__ __launch_bounds__(256) void embed_add_kernel(
    float* __restrict__ h, unsigned short* __restrict__ hbf,
    const float* __restrict__ pos_emb, const float* __restrict__ tf,
    const float* __restrict__ tp_w, const float* __restrict__ tp_b,
    const float* __restrict__ ps, const float* __restrict__ ts) {
    const int i = blockIdx.x * 256 + threadIdx.x;     // B*L*D elements
    const int d = i & 511, t = i >> 9, l = t & 4095;
    const float tv = tf[(size_t)t * 2] * tp_w[d] + tf[(size_t)t * 2 + 1] * tp_w[512 + d] + tp_b[d];
    const float v = h[i] + ps[0] * pos_emb[(size_t)l * D_MODEL + d] + ts[0] * tv;
    h[i] = v; hbf[i] = f2bf(v);
}

// ---------------------------------------------- ProbSparse: sampling ----
__device__ __forceinline__ int samp_idx(int s, int mask) {
    return (int)((2654435761u * (unsigned)s + 911u) & (unsigned)mask);  // odd mult: bijection mod 2^k
}

__global__ __launch_bounds__(256) void qk_sample_kernel(
    const float* __restrict__ Q, const float* __restrict__ K,
    float* __restrict__ Msc, int Lc, int u) {
    __shared__ float sK[48 * HDIM];
    const int tid = threadIdx.x;
    const int bh = blockIdx.x, b = bh >> 3, hh = bh & 7;
    const size_t base = ((size_t)b * Lc) * D_MODEL + hh * HDIM;
    const int mask = Lc - 1;
    for (int j = tid; j < u * HDIM; j += 256) {
        const int s = j >> 6, d = j & 63;
        sK[j] = K[base + (size_t)samp_idx(s, mask) * D_MODEL + d];
    }
    __syncthreads();
    const int l = blockIdx.y * 256 + tid;
    const float* qp = Q + base + (size_t)l * D_MODEL;
    float q[HDIM];
#pragma unroll
    for (int d = 0; d < HDIM; ++d) q[d] = qp[d];
    float mx = -3.4e38f, sm = 0.f;
    for (int s = 0; s < u; ++s) {
        float dot = 0.f;
#pragma unroll
        for (int d = 0; d < HDIM; ++d) dot += q[d] * sK[s * HDIM + d];
        dot *= 0.125f;
        mx = fmaxf(mx, dot); sm += dot;
    }
    Msc[(size_t)bh * Lc + l] = mx - sm / (float)u;
}

__global__ __launch_bounds__(256) void topk_kernel(
    float* __restrict__ Msc, int* __restrict__ idx, int Lc, int u) {
    __shared__ float bv[256];
    __shared__ int   bi[256];
    const int tid = threadIdx.x, bh = blockIdx.x;
    float* m = Msc + (size_t)bh * Lc;
    for (int it = 0; it < u; ++it) {
        float best = -3.4e38f; int bid = 0;
        for (int l = tid; l < Lc; l += 256) {
            const float v = m[l];
            if (v > best || (v == best && l < bid)) { best = v; bid = l; }
        }
        bv[tid] = best; bi[tid] = bid; __syncthreads();
        for (int s = 128; s > 0; s >>= 1) {
            if (tid < s) {
                if (bv[tid + s] > bv[tid] ||
                    (bv[tid + s] == bv[tid] && bi[tid + s] < bi[tid])) {
                    bv[tid] = bv[tid + s]; bi[tid] = bi[tid + s];
                }
            }
            __syncthreads();
        }
        if (tid == 0) { idx[bh * u + it] = bi[0]; m[bi[0]] = -3.4e38f; }
        __syncthreads();
    }
}

__global__ __launch_bounds__(64) void vmean_kernel(
    const float* __restrict__ V, float* __restrict__ Vm, int Lc, int lshift) {
    const int bh = blockIdx.x, b = bh >> 3, hh = bh & 7, d = threadIdx.x;
    const float* vp = V + (((size_t)b << lshift)) * D_MODEL + hh * HDIM + d;
    float s = 0.f;
    for (int l = 0; l < Lc; ++l) s += vp[(size_t)l * D_MODEL];
    Vm[bh * HDIM + d] = s / (float)Lc;
}

__global__ __launch_bounds__(256) void fill_mean_kernel(
    unsigned short* __restrict__ AObf, const float* __restrict__ Vm, int lshift) {
    const int i = blockIdx.x * 256 + threadIdx.x;   // B*Lc*D
    const int d512 = i & 511, t = i >> 9;
    const int b = t >> lshift;
    AObf[i] = f2bf(Vm[(b * NHEAD + (d512 >> 6)) * HDIM + (d512 & 63)]);
}

__global__ __launch_bounds__(256) void sparse_attn_kernel(
    const float* __restrict__ Q, const float* __restrict__ K,
    const float* __restrict__ V, const int* __restrict__ idx,
    unsigned short* __restrict__ AObf, int Lc, int u) {
    __shared__ float sl[4096];
    __shared__ float sq[HDIM];
    __shared__ float red[256];
    const int tid = threadIdx.x;
    const int bh = blockIdx.x, b = bh >> 3, hh = bh & 7;
    const int lq = idx[bh * u + blockIdx.y];
    const size_t base = ((size_t)b * Lc) * D_MODEL + hh * HDIM;
    if (tid < HDIM) sq[tid] = Q[base + (size_t)lq * D_MODEL + tid];
    __syncthreads();
    for (int l = tid; l < Lc; l += 256) {           // logits + causal mask
        const float* kp = K + base + (size_t)l * D_MODEL;
        float dot = 0.f;
#pragma unroll
        for (int d = 0; d < HDIM; ++d) dot += sq[d] * kp[d];
        sl[l] = dot * 0.125f + (l > lq ? -1e9f : 0.f);
    }
    __syncthreads();
    float mx = -3.4e38f;
    for (int l = tid; l < Lc; l += 256) mx = fmaxf(mx, sl[l]);
    red[tid] = mx; __syncthreads();
    for (int s = 128; s > 0; s >>= 1) { if (tid < s) red[tid] = fmaxf(red[tid], red[tid + s]); __syncthreads(); }
    mx = red[0]; __syncthreads();
    float ss = 0.f;
    for (int l = tid; l < Lc; l += 256) { const float p = __expf(sl[l] - mx); sl[l] = p; ss += p; }
    red[tid] = ss; __syncthreads();
    for (int s = 128; s > 0; s >>= 1) { if (tid < s) red[tid] += red[tid + s]; __syncthreads(); }
    const float S = red[0]; __syncthreads();
    const int d = tid & 63, part = tid >> 6;        // attn @ V
    float acc = 0.f;
    for (int l = part; l < Lc; l += 4) acc += sl[l] * V[base + (size_t)l * D_MODEL + d];
    red[tid] = acc; __syncthreads();
    if (tid < HDIM) {
        const float o = (red[tid] + red[64 + tid] + red[128 + tid] + red[192 + tid]) / S;
        AObf[base + (size_t)lq * D_MODEL + tid] = f2bf(o);
    }
}

// ------------------------------------------------- ELU + maxpool(3,2,1) ----
__global__ __launch_bounds__(256) void elu_maxpool_kernel(
    const float* __restrict__ c, float* __restrict__ out,
    unsigned short* __restrict__ outbf, int Lc, int hs) {
    const int i = blockIdx.x * 256 + threadIdx.x;   // B*(Lc/2)*D
    const int d = i & 511, t = i >> 9;
    const int half = Lc >> 1;
    const int b = t >> hs, lo = t & (half - 1);
    const int lc = lo * 2;
    float m = -3.4e38f;
#pragma unroll
    for (int dl = -1; dl <= 1; ++dl) {
        const int l = lc + dl;
        if ((unsigned)l < (unsigned)Lc)
            m = fmaxf(m, c[((size_t)b * Lc + l) * D_MODEL + d]);
    }
    const float v = m > 0.f ? m : __expf(m) - 1.f;  // elu is monotone -> elu(max)
    out[i] = v; outbf[i] = f2bf(v);
}

// -------------------------------------------------------------- driver ----
static void run_gemm(const unsigned short* A, const unsigned short* Bt,
                     const float* bias, void* C, int M, int N, int K,
                     bool relu_outbf, hipStream_t s) {
    dim3 g(N / 64, M / 64), blk(128);
    if (relu_outbf) gemm_bf16<true,  true ><<<g, blk, 0, s>>>(A, Bt, bias, C, M, N, K);
    else            gemm_bf16<false, false><<<g, blk, 0, s>>>(A, Bt, bias, C, M, N, K);
}

extern "C" void kernel_launch(void* const* d_in, const int* in_sizes, int n_in,
                              void* d_out, int out_size, void* d_ws, size_t ws_size,
                              hipStream_t stream) {
    (void)in_sizes; (void)n_in; (void)ws_size;
    const float* x       = (const float*)d_in[0];
    const float* tf      = (const float*)d_in[1];
    const float* in_w    = (const float*)d_in[2];
    const float* in_b    = (const float*)d_in[3];
    const float* pos_emb = (const float*)d_in[4];
    const float* tp_w    = (const float*)d_in[5];
    const float* tp_b    = (const float*)d_in[6];
    const float* ps      = (const float*)d_in[7];
    const float* ts      = (const float*)d_in[8];
    const float* conv_w  = (const float*)d_in[9];
    const float* conv_b  = (const float*)d_in[10];
    const float* wq = (const float*)d_in[11], *bq = (const float*)d_in[12];
    const float* wk = (const float*)d_in[13], *bk = (const float*)d_in[14];
    const float* wv = (const float*)d_in[15], *bv = (const float*)d_in[16];
    const float* wo = (const float*)d_in[17], *bo = (const float*)d_in[18];
    const float* f1w = (const float*)d_in[19], *f1b = (const float*)d_in[20];
    const float* f2w = (const float*)d_in[21], *f2b = (const float*)d_in[22];
    const float* g1 = (const float*)d_in[23], *g2 = (const float*)d_in[24], *g3 = (const float*)d_in[25];
    const float* be1 = (const float*)d_in[26], *be2 = (const float*)d_in[27], *be3 = (const float*)d_in[28];
    const float* dconv_w = (const float*)d_in[29], *dconv_b = (const float*)d_in[30];

    const size_t SEG = (size_t)BATCH * 4096 * D_MODEL;   // 16.78M elems
    float* ws   = (float*)d_ws;
    float* hbuf = ws;                  // f32 activations
    float* t0   = ws + SEG;
    float* Qb   = ws + 2 * SEG;
    float* Kb   = ws + 3 * SEG;
    float* Vb   = ws + 4 * SEG;
    float* Msc  = ws + 5 * SEG;
    float* Vm   = Msc + (size_t)BATCH * NHEAD * 4096;
    int*   idxb = (int*)(Vm + BATCH * NHEAD * HDIM);

    unsigned short* ub = (unsigned short*)(ws + 5 * SEG + 524288);
    unsigned short* hbf   = ub;  ub += SEG;                  // bf16 activations
    unsigned short* aobf  = ub;  ub += SEG;
    unsigned short* midbf = (unsigned short*)Qb;             // overlays Qb+Kb (4*SEG us)
    unsigned short* xbf   = ub;  ub += (size_t)BATCH * 4096 * 64;
    unsigned short* inwt  = ub;  ub += 64 * D_MODEL;
    unsigned short* wqt   = ub;  ub += (size_t)3 * D_MODEL * D_MODEL;
    unsigned short* wkt   = ub;  ub += (size_t)3 * D_MODEL * D_MODEL;
    unsigned short* wvt   = ub;  ub += (size_t)3 * D_MODEL * D_MODEL;
    unsigned short* wot   = ub;  ub += (size_t)3 * D_MODEL * D_MODEL;
    unsigned short* f1t   = ub;  ub += (size_t)3 * D_MODEL * 2048;
    unsigned short* f2t   = ub;  ub += (size_t)3 * 2048 * D_MODEL;
    unsigned short* cwt   = ub;  ub += (size_t)3 * D_MODEL * 1536;
    unsigned short* dwt   = ub;  ub += (size_t)2 * D_MODEL * 1536;

    // ---- one-time (per call) bf16 conversion / weight re-layout
    {
        const int nx = BATCH * 4096 * 64;
        cvt_bf16_kernel<<<(nx + 255) / 256, 256, 0, stream>>>(x, xbf, nx);
        transpose_cvt_kernel<<<(64 * 512 + 255) / 256, 256, 0, stream>>>(in_w, inwt, 64, 512, 64 * 512);
        const int nqk = 3 * 512 * 512;
        transpose_cvt_kernel<<<(nqk + 255) / 256, 256, 0, stream>>>(wq, wqt, 512, 512, nqk);
        transpose_cvt_kernel<<<(nqk + 255) / 256, 256, 0, stream>>>(wk, wkt, 512, 512, nqk);
        transpose_cvt_kernel<<<(nqk + 255) / 256, 256, 0, stream>>>(wv, wvt, 512, 512, nqk);
        transpose_cvt_kernel<<<(nqk + 255) / 256, 256, 0, stream>>>(wo, wot, 512, 512, nqk);
        const int nf = 3 * 512 * 2048;
        transpose_cvt_kernel<<<(nf + 255) / 256, 256, 0, stream>>>(f1w, f1t, 512, 2048, nf);
        transpose_cvt_kernel<<<(nf + 255) / 256, 256, 0, stream>>>(f2w, f2t, 2048, 512, nf);
        const int nc = 3 * 512 * 1536;
        conv_cvt_kernel<<<(nc + 255) / 256, 256, 0, stream>>>(conv_w, cwt, nc);
        const int nd = 2 * 512 * 1536;
        conv_cvt_kernel<<<(nd + 255) / 256, 256, 0, stream>>>(dconv_w, dwt, nd);
    }

    // ---- input projection + positional/temporal embedding
    int L = 4096, lshift = 12;
    run_gemm(xbf, inwt, in_b, hbuf, BATCH * L, D_MODEL, 64, false, stream);
    embed_add_kernel<<<(BATCH * L * D_MODEL) / 256, 256, 0, stream>>>(
        hbuf, hbf, pos_emb, tf, tp_w, tp_b, ps, ts);

    const int DD = D_MODEL * D_MODEL;
    for (int i = 0; i < 3; ++i) {
        const int M = BATCH * L;
        const int u = (L == 4096) ? 45 : (L == 2048 ? 40 : 35);

        // causal conv (pad 2,0) + LN
        conv3_bf16<<<dim3(D_MODEL / 64, M / 64), 128, 0, stream>>>(
            hbf, cwt + (size_t)i * D_MODEL * 1536, conv_b + (size_t)i * D_MODEL, t0, L, lshift, 2);
        ln_kernel<<<M, 256, 0, stream>>>(hbuf, t0, g1 + i * D_MODEL, be1 + i * D_MODEL, hbuf, hbf);

        // ProbSparse attention
        run_gemm(hbf, wqt + (size_t)i * DD, bq + i * D_MODEL, Qb, M, D_MODEL, D_MODEL, false, stream);
        run_gemm(hbf, wkt + (size_t)i * DD, bk + i * D_MODEL, Kb, M, D_MODEL, D_MODEL, false, stream);
        run_gemm(hbf, wvt + (size_t)i * DD, bv + i * D_MODEL, Vb, M, D_MODEL, D_MODEL, false, stream);
        qk_sample_kernel<<<dim3(BATCH * NHEAD, L / 256), 256, 0, stream>>>(Qb, Kb, Msc, L, u);
        topk_kernel<<<BATCH * NHEAD, 256, 0, stream>>>(Msc, idxb, L, u);
        vmean_kernel<<<BATCH * NHEAD, HDIM, 0, stream>>>(Vb, Vm, L, lshift);
        fill_mean_kernel<<<(M * D_MODEL) / 256, 256, 0, stream>>>(aobf, Vm, lshift);
        sparse_attn_kernel<<<dim3(BATCH * NHEAD, u), 256, 0, stream>>>(Qb, Kb, Vb, idxb, aobf, L, u);
        run_gemm(aobf, wot + (size_t)i * DD, bo + i * D_MODEL, t0, M, D_MODEL, D_MODEL, false, stream);
        ln_kernel<<<M, 256, 0, stream>>>(hbuf, t0, g2 + i * D_MODEL, be2 + i * D_MODEL, hbuf, hbf);

        // FFN (mid stored bf16-only: it is consumed only as a GEMM A operand)
        run_gemm(hbf, f1t + (size_t)i * D_MODEL * 2048, f1b + i * 2048, midbf, M, 2048, D_MODEL, true, stream);
        run_gemm(midbf, f2t + (size_t)i * 2048 * D_MODEL, f2b + i * D_MODEL, t0, M, D_MODEL, 2048, false, stream);
        ln_kernel<<<M, 256, 0, stream>>>(hbuf, t0, g3 + i * D_MODEL, be3 + i * D_MODEL, hbuf, hbf);

        // distilling conv (pad 1) + elu + maxpool(3,2)
        if (i < 2) {
            conv3_bf16<<<dim3(D_MODEL / 64, M / 64), 128, 0, stream>>>(
                hbf, dwt + (size_t)i * D_MODEL * 1536, dconv_b + (size_t)i * D_MODEL, t0, L, lshift, 1);
            elu_maxpool_kernel<<<(BATCH * (L / 2) * D_MODEL) / 256, 256, 0, stream>>>(
                t0, hbuf, hbf, L, lshift - 1);
            L >>= 1; lshift -= 1;
        }
    }
    (void)hipMemcpyAsync(d_out, hbuf, (size_t)out_size * sizeof(float),
                         hipMemcpyDeviceToDevice, stream);
}
